// xLSTMModel_58995670777868
// MI455X (gfx1250) — compile-verified, matrix operand order checked by reading, not run on hardware
//
#include <hip/hip_runtime.h>
#include <hip/hip_bf16.h>

// ---------------------------------------------------------------------------
// xLSTM forward for MI455X (gfx1250, wave32): bf16 WMMA + async-to-LDS copies
// ---------------------------------------------------------------------------

#define B_  256
#define S_  100
#define I_  128
#define H_  1024
#define O_  128
#define L_  2
#define U_  1344
#define KC_ 4
#define BS_ (B_ * S_)   // 25600

typedef __attribute__((ext_vector_type(16))) __bf16 v16bf_t;
typedef __attribute__((ext_vector_type(8)))  float  v8f_t;

static __device__ __forceinline__ __bf16 f2bf(float f) { return (__bf16)f; }

// generic (flat) shared pointer -> 32-bit LDS byte offset
static __device__ __forceinline__ unsigned lds_addr(const void* p) {
    return (unsigned)(unsigned long long)(__attribute__((address_space(3))) const char*)p;
}

// CDNA5 async copy: LDS[ldsOff..+15] = MEM[gaddr..+15]   (tracked by ASYNCcnt)
static __device__ __forceinline__ void async_b128(unsigned ldsOff, const void* g) {
    asm volatile("global_load_async_to_lds_b128 %0, %1, off"
                 :: "v"(ldsOff), "v"((unsigned long long)g) : "memory");
}
#define WAIT_ASYNC(n) asm volatile("s_wait_asynccnt %0" :: "i"(n) : "memory")

// ---------------------------------------------------------------------------
// bf16 WMMA GEMM: C[M,N] = A[M,K] @ W[N,K]^T (+bias[N]) (+Cres)
// A, W are bf16. 256 threads = 8 waves. Block tile 128x128, K step 32.
// Wave grid 4(M) x 2(N); each wave: 32x64 = 2x4 accumulators (8 WMMA/K-step).
// Double-buffered LDS fed by global_load_async_to_lds_b128.
// ---------------------------------------------------------------------------
#define GT_BM 128
#define GT_BN 128
#define GT_BK 32

__global__ __launch_bounds__(256) void gemm_bf16_wmma(
    const __bf16* __restrict__ A, int lda,      // [M,K] bf16
    const __bf16* __restrict__ W,               // [N,K] bf16
    const float* __restrict__ bias,             // [N] or nullptr
    const float* __restrict__ Cres, int ldres,  // residual or nullptr
    float* __restrict__ C, int ldc,
    int M, int N, int K)
{
    __shared__ __bf16 As[2][GT_BM * GT_BK];
    __shared__ __bf16 Ws[2][GT_BN * GT_BK];

    const int tid   = threadIdx.x;
    const int lane  = tid & 31;
    const int wid   = tid >> 5;
    const int waveM = wid & 3;       // 4 x 32 rows
    const int waveN = wid >> 2;      // 2 x 64 cols
    const int m0 = blockIdx.y * GT_BM;
    const int n0 = blockIdx.x * GT_BN;

    v8f_t acc[2][4];
    #pragma unroll
    for (int i = 0; i < 2; ++i)
        #pragma unroll
        for (int j = 0; j < 4; ++j) acc[i][j] = (v8f_t){};

    // per K-tile: A 128x32 bf16 = 512 b128 chunks, W same; 4 asyncs/thread
    const int nk = K / GT_BK;
    {
        #pragma unroll
        for (int i = 0; i < 2; ++i) {
            const int ch = tid * 2 + i;
            const int row = ch >> 2, col = (ch & 3) * 8;
            async_b128(lds_addr(&As[0][row * GT_BK + col]),
                       A + (size_t)(m0 + row) * lda + col);
            async_b128(lds_addr(&Ws[0][row * GT_BK + col]),
                       W + (size_t)(n0 + row) * K + col);
        }
    }

    for (int k = 0; k < nk; ++k) {
        if (k + 1 < nk) {
            const int kn = (k + 1) * GT_BK;
            const int bn = (k + 1) & 1;
            #pragma unroll
            for (int i = 0; i < 2; ++i) {
                const int ch = tid * 2 + i;
                const int row = ch >> 2, col = (ch & 3) * 8;
                async_b128(lds_addr(&As[bn][row * GT_BK + col]),
                           A + (size_t)(m0 + row) * lda + kn + col);
                async_b128(lds_addr(&Ws[bn][row * GT_BK + col]),
                           W + (size_t)(n0 + row) * K + kn + col);
            }
            WAIT_ASYNC(4);   // 4 outstanding = next tile; current tile done
        } else {
            WAIT_ASYNC(0);
        }
        __syncthreads();

        const __bf16* as = As[k & 1];
        const __bf16* ws = Ws[k & 1];
        // ISA 7.12.2 16-bit operand layout: lanes 0-15 hold K 0-7 / 16-23,
        // lanes 16-31 hold K 8-15 / 24-31; row/col = lane % 16.
        const int r  = lane & 15;
        const int kb = (lane < 16) ? 0 : 8;

        v16bf_t af[2];
        #pragma unroll
        for (int t = 0; t < 2; ++t) {
            const __bf16* ap = as + (waveM * 32 + t * 16 + r) * GT_BK;
            #pragma unroll
            for (int e = 0; e < 8; ++e) {
                af[t][e]     = ap[kb + e];
                af[t][8 + e] = ap[kb + 16 + e];
            }
        }
        v16bf_t bf[4];
        #pragma unroll
        for (int t = 0; t < 4; ++t) {
            const __bf16* bp = ws + (waveN * 64 + t * 16 + r) * GT_BK;
            #pragma unroll
            for (int e = 0; e < 8; ++e) {
                bf[t][e]     = bp[kb + e];
                bf[t][8 + e] = bp[kb + 16 + e];
            }
        }
        #pragma unroll
        for (int mi = 0; mi < 2; ++mi)
            #pragma unroll
            for (int ni = 0; ni < 4; ++ni)
                acc[mi][ni] = __builtin_amdgcn_wmma_f32_16x16x32_bf16(
                    false, af[mi], false, bf[ni], (short)0, acc[mi][ni],
                    false, false);
        __syncthreads();
    }

    // epilogue: elem rr -> row rr (+8 for lanes 16-31), col = lane%16
    const int colL = lane & 15;
    const int rofs = (lane >> 4) * 8;
    #pragma unroll
    for (int mi = 0; mi < 2; ++mi) {
        #pragma unroll
        for (int ni = 0; ni < 4; ++ni) {
            const int cc = n0 + waveN * 64 + ni * 16 + colL;
            #pragma unroll
            for (int rr = 0; rr < 8; ++rr) {
                const int row = m0 + waveM * 32 + mi * 16 + rr + rofs;
                float v = acc[mi][ni][rr];
                if (bias) v += bias[cc];
                if (Cres) v += Cres[(size_t)row * ldres + cc];
                C[(size_t)row * ldc + cc] = v;
            }
        }
    }
}

// ---------------------------------------------------------------------------
// LayerNorm (weight only, eps=1e-5). 256 threads per row of H=1024.
// mode 0: row r at in + r*H
// mode 1: in is ys[S,B,H]; row r=(b*S+s) read at (s*B+b)*H ; addRes adds out[]
// mode 2: rows=B; row r at (r*S + S-1)*H (last token)
// Optional bf16 secondary output (GEMM operand feed).
// ---------------------------------------------------------------------------
__global__ __launch_bounds__(256) void layernorm_k(
    const float* __restrict__ in, const float* __restrict__ w,
    float* __restrict__ out, __bf16* __restrict__ out_bf,
    int mode, int addRes)
{
    const int r = blockIdx.x;
    size_t ibase;
    if (mode == 0)      ibase = (size_t)r * H_;
    else if (mode == 1) { int b = r / S_, s = r % S_; ibase = ((size_t)s * B_ + b) * H_; }
    else                ibase = ((size_t)r * S_ + (S_ - 1)) * H_;
    const size_t obase = (size_t)r * H_;

    const int tid = threadIdx.x;
    float x[4];
    float sum = 0.f, sq = 0.f;
    #pragma unroll
    for (int e = 0; e < 4; ++e) {
        x[e] = in[ibase + tid + e * 256];
        sum += x[e];
        sq  += x[e] * x[e];
    }
    __shared__ float rs[256], rq[256];
    rs[tid] = sum; rq[tid] = sq;
    __syncthreads();
    for (int st = 128; st > 0; st >>= 1) {
        if (tid < st) { rs[tid] += rs[tid + st]; rq[tid] += rq[tid + st]; }
        __syncthreads();
    }
    const float mu  = rs[0] * (1.0f / H_);
    const float var = rq[0] * (1.0f / H_) - mu * mu;
    const float inv = rsqrtf(var + 1e-5f);
    #pragma unroll
    for (int e = 0; e < 4; ++e) {
        const int j = tid + e * 256;
        float v = (x[e] - mu) * inv * w[j];
        if (addRes) v += out[obase + j];
        out[obase + j] = v;
        if (out_bf) out_bf[obase + j] = f2bf(v);
    }
}

// ---------------------------------------------------------------------------
// Depthwise causal conv1d (K=4) + SiLU -> bf16 (GEMM operand only)
// ---------------------------------------------------------------------------
__global__ __launch_bounds__(256) void conv_silu_k(
    const float* __restrict__ xn, const float* __restrict__ w,  // [H,KC]
    const float* __restrict__ bias, __bf16* __restrict__ xcbf)
{
    const size_t idx = (size_t)blockIdx.x * 256 + threadIdx.x;
    const int h = (int)(idx % H_);
    const int s = (int)((idx / H_) % S_);
    const int b = (int)(idx / ((size_t)H_ * S_));
    float acc = bias[h];
    #pragma unroll
    for (int k = 0; k < KC_; ++k) {
        const int sp = s - (KC_ - 1) + k;
        if (sp >= 0)
            acc += xn[((size_t)b * S_ + sp) * H_ + h] * w[h * KC_ + k];
    }
    xcbf[idx] = f2bf(acc / (1.f + expf(-acc)));   // silu
}

// ---------------------------------------------------------------------------
// fp32 -> bf16 flat convert (weights / model input, once per call)
// ---------------------------------------------------------------------------
__global__ __launch_bounds__(256) void convert_bf16_k(
    const float* __restrict__ src, __bf16* __restrict__ dst)
{
    const size_t idx = (size_t)blockIdx.x * 256 + threadIdx.x;
    dst[idx] = f2bf(src[idx]);
}

// Transpose R [H,4H] fp32 -> Rt [4H,H] bf16
__global__ __launch_bounds__(256) void r_transpose_bf16_k(
    const float* __restrict__ R, __bf16* __restrict__ Rt)
{
    const size_t idx = (size_t)blockIdx.x * 256 + threadIdx.x;  // over 4H*H
    const int j = (int)(idx / H_);
    const int k = (int)(idx % H_);
    Rt[idx] = f2bf(R[(size_t)k * (4 * H_) + j]);
}

// GELU(exact)-gated FFN elementwise: t[r,u] = gelu(gv[r,u]) * gv[r,U+u]
__global__ __launch_bounds__(256) void gelu_gate_k(
    const float* __restrict__ gv, __bf16* __restrict__ t)
{
    const size_t idx = (size_t)blockIdx.x * 256 + threadIdx.x;  // over BS*U
    const size_t r = idx / U_;
    const int    u = (int)(idx % U_);
    const float g = gv[r * (2 * U_) + u];
    const float v = gv[r * (2 * U_) + U_ + u];
    t[idx] = f2bf(0.5f * g * (1.f + erff(g * 0.70710678118654752f)) * v);
}

// ---------------------------------------------------------------------------
// Fused sLSTM step (one launch per timestep). 128 threads = 4 waves = 4 gates.
// Wave g: raw_g = y_prev @ R_g^T for a 32(batch) x 16(h) tile via WMMA with
// async-to-LDS operand staging; gates combined through LDS; stabilized
// exponential-gating update. Grid (H/16, B/32). Wx holds x-proj + cell bias.
// ---------------------------------------------------------------------------
__global__ __launch_bounds__(128) void slstm_step_k(
    const float*  __restrict__ Wx,       // [BS,4H], row = b*S + s
    const __bf16* __restrict__ Rt,       // [4H,H]
    const __bf16* __restrict__ ybf_prev, // [B,H]
    float*  __restrict__ ys_s,           // f32 ys + s*B*H (for group norm)
    __bf16* __restrict__ ybf_new,        // [B,H]
    float* __restrict__ cst, float* __restrict__ nst, float* __restrict__ mst,
    int s)
{
    const int lane = threadIdx.x & 31;
    const int gate = threadIdx.x >> 5;
    const int n0 = blockIdx.x * 16;
    const int m0 = blockIdx.y * 32;

    __shared__ __bf16 Ys[32 * 32];
    __shared__ __bf16 Rs[4][16 * 32];
    __shared__ float  rawS[4][32 * 16];

    v8f_t acc0 = {};
    v8f_t acc1 = {};

    const int yRow = threadIdx.x >> 2;          // 0..31
    const int yCol = (threadIdx.x & 3) * 8;
    const int rRow = lane >> 1;                 // 0..15
    const int rCol = (lane & 1) * 16;

    for (int k0 = 0; k0 < H_; k0 += 32) {
        // y tile 32x32 bf16: 1 b128/thread; R tile 16x32/gate: 2 b128/lane
        async_b128(lds_addr(&Ys[yRow * 32 + yCol]),
                   ybf_prev + (size_t)(m0 + yRow) * H_ + k0 + yCol);
        #pragma unroll
        for (int i = 0; i < 2; ++i)
            async_b128(lds_addr(&Rs[gate][rRow * 32 + rCol + i * 8]),
                       Rt + (size_t)(gate * H_ + n0 + rRow) * H_ + k0 + rCol + i * 8);
        WAIT_ASYNC(0);
        __syncthreads();

        const int r  = lane & 15;
        const int kb = (lane < 16) ? 0 : 8;
        v16bf_t bfrag;
        {
            const __bf16* bp = Rs[gate] + r * 32;
            #pragma unroll
            for (int e = 0; e < 8; ++e) {
                bfrag[e]     = bp[kb + e];
                bfrag[8 + e] = bp[kb + 16 + e];
            }
        }
        v16bf_t a0, a1;
        {
            const __bf16* a0p = Ys + r * 32;
            const __bf16* a1p = Ys + (16 + r) * 32;
            #pragma unroll
            for (int e = 0; e < 8; ++e) {
                a0[e]     = a0p[kb + e];
                a0[8 + e] = a0p[kb + 16 + e];
                a1[e]     = a1p[kb + e];
                a1[8 + e] = a1p[kb + 16 + e];
            }
        }
        acc0 = __builtin_amdgcn_wmma_f32_16x16x32_bf16(false, a0, false, bfrag,
                                                       (short)0, acc0, false, false);
        acc1 = __builtin_amdgcn_wmma_f32_16x16x32_bf16(false, a1, false, bfrag,
                                                       (short)0, acc1, false, false);
        __syncthreads();
    }

    {   // stage raw_g through LDS
        const int colL = lane & 15;
        const int rofs = (lane >> 4) * 8;
        #pragma unroll
        for (int rr = 0; rr < 8; ++rr) {
            rawS[gate][(rr + rofs) * 16 + colL]      = acc0[rr];
            rawS[gate][(16 + rr + rofs) * 16 + colL] = acc1[rr];
        }
    }
    __syncthreads();

    // state update: 512 elements, 128 threads x 4
    for (int i = threadIdx.x; i < 32 * 16; i += 128) {
        const int row = i >> 4;
        const int col = i & 15;
        const int b  = m0 + row;
        const int hh = n0 + col;
        const size_t wb = ((size_t)b * S_ + s) * (4 * H_) + hh;
        const float ir  = rawS[0][i] + Wx[wb];
        const float fr  = rawS[1][i] + Wx[wb + H_];
        const float zr  = rawS[2][i] + Wx[wb + 2 * H_];
        const float orr = rawS[3][i] + Wx[wb + 3 * H_];

        const size_t si = (size_t)b * H_ + hh;
        const float c = cst[si], n = nst[si], m = mst[si];

        const float lsig = fminf(fr, 0.f) - log1pf(expf(-fabsf(fr)));
        const float logfplusm = m + lsig;
        const float mnew = (n == 0.0f) ? ir : fmaxf(ir, logfplusm);
        const float ig = expf(ir - mnew);
        const float fg = expf(logfplusm - mnew);
        const float cnew = fg * c + ig * tanhf(zr);
        const float nnew = fg * n + ig;
        const float yn = (1.f / (1.f + expf(-orr))) * cnew / nnew;

        cst[si] = cnew; nst[si] = nnew; mst[si] = mnew;
        ys_s[si] = yn;
        ybf_new[si] = f2bf(yn);
    }
}

// ---------------------------------------------------------------------------
// Host orchestration
// ---------------------------------------------------------------------------
static inline void launch_gemm(const __bf16* A, int lda, const __bf16* W,
                               const float* bias, const float* Cres, int ldres,
                               float* C, int ldc, int M, int N, int K,
                               hipStream_t stream)
{
    dim3 grid(N / GT_BN, M / GT_BM);
    gemm_bf16_wmma<<<grid, 256, 0, stream>>>(A, lda, W, bias, Cres, ldres,
                                             C, ldc, M, N, K);
}
static inline void launch_convert(const float* s, __bf16* d, size_t n,
                                  hipStream_t stream)
{
    convert_bf16_k<<<n / 256, 256, 0, stream>>>(s, d);
}

extern "C" void kernel_launch(void* const* d_in, const int* in_sizes, int n_in,
                              void* d_out, int out_size, void* d_ws, size_t ws_size,
                              hipStream_t stream)
{
    (void)in_sizes; (void)n_in; (void)out_size; (void)ws_size;

    const float* x      = (const float*)d_in[0];   // [B,S,I]
    const float* emb_w  = (const float*)d_in[1];   // [H,I]
    const float* emb_b  = (const float*)d_in[2];
    const float* conv_w = (const float*)d_in[3];   // [L,H,KC]
    const float* conv_b = (const float*)d_in[4];
    const float* Wi     = (const float*)d_in[5];   // [L,H,H]
    const float* Wf     = (const float*)d_in[6];
    const float* Wz     = (const float*)d_in[7];
    const float* Wo     = (const float*)d_in[8];
    const float* R      = (const float*)d_in[9];   // [L,H,4H]
    const float* cell_b = (const float*)d_in[10];  // [L,4H]
    const float* gn_w   = (const float*)d_in[11];
    const float* ln1_w  = (const float*)d_in[12];
    const float* ln2_w  = (const float*)d_in[13];
    const float* ff_up  = (const float*)d_in[14];  // [L,2U,H]
    const float* ff_dn  = (const float*)d_in[15];  // [L,H,U]
    const float* post_w = (const float*)d_in[16];
    const float* fc_w   = (const float*)d_in[17];  // [O,H]
    const float* fc_b   = (const float*)d_in[18];
    float* out = (float*)d_out;                    // [B,O]

    // ---- workspace layout (16B-aligned segments) ----
    float* ws = (float*)d_ws;
    size_t off = 0;
    auto allocF = [&](size_t n) { float* p = ws + off; off += (n + 63) & ~(size_t)63; return p; };
    auto allocB = [&](size_t n) { return (__bf16*)allocF((n + 1) / 2); };

    float* h   = allocF((size_t)BS_ * H_);
    float* xn  = allocF((size_t)BS_ * H_);
    float* ys  = allocF((size_t)BS_ * H_);          // [S,B,H]
    float* Wx  = allocF((size_t)BS_ * 4 * H_);      // [BS,4H]; FFN gv reuses
    float* cst = allocF((size_t)B_ * H_);
    float* nst = allocF((size_t)B_ * H_);
    float* mst = allocF((size_t)B_ * H_);
    float* hn  = allocF((size_t)B_ * H_);

    __bf16* xbf  = allocB((size_t)BS_ * I_);
    __bf16* xnbf = allocB((size_t)BS_ * H_);
    __bf16* xcbf = allocB((size_t)BS_ * H_);
    __bf16* tgbf = allocB((size_t)BS_ * U_);
    __bf16* hnbf = allocB((size_t)B_ * H_);
    __bf16* ybf0 = allocB((size_t)B_ * H_);
    __bf16* ybf1 = allocB((size_t)B_ * H_);
    __bf16* ybz  = allocB((size_t)B_ * H_);
    __bf16* Rt   = allocB((size_t)4 * H_ * H_);     // [4H,H]
    __bf16* embw_bf = allocB((size_t)H_ * I_);
    __bf16* fcw_bf  = allocB((size_t)O_ * H_);
    __bf16* gwbf[L_][4];
    __bf16* upbf[L_];
    __bf16* dnbf[L_];
    for (int l = 0; l < L_; ++l) {
        for (int g = 0; g < 4; ++g) gwbf[l][g] = allocB((size_t)H_ * H_);
        upbf[l] = allocB((size_t)2 * U_ * H_);
        dnbf[l] = allocB((size_t)H_ * U_);
    }
    float* gv = Wx;                                 // [BS,2U] (reuse)

    // ---- one-time per-call bf16 weight/input conversion ----
    launch_convert(x, xbf, (size_t)BS_ * I_, stream);
    launch_convert(emb_w, embw_bf, (size_t)H_ * I_, stream);
    launch_convert(fc_w, fcw_bf, (size_t)O_ * H_, stream);
    const float* gws[L_][4];
    for (int l = 0; l < L_; ++l) {
        gws[l][0] = Wi + (size_t)l * H_ * H_;
        gws[l][1] = Wf + (size_t)l * H_ * H_;
        gws[l][2] = Wz + (size_t)l * H_ * H_;
        gws[l][3] = Wo + (size_t)l * H_ * H_;
        for (int g = 0; g < 4; ++g)
            launch_convert(gws[l][g], gwbf[l][g], (size_t)H_ * H_, stream);
        launch_convert(ff_up + (size_t)l * 2 * U_ * H_, upbf[l],
                       (size_t)2 * U_ * H_, stream);
        launch_convert(ff_dn + (size_t)l * H_ * U_, dnbf[l],
                       (size_t)H_ * U_, stream);
    }

    // embedding: h = x @ emb_w^T + emb_b
    launch_gemm(xbf, I_, embw_bf, emb_b, nullptr, 0, h, H_, BS_, H_, I_, stream);

    for (int l = 0; l < L_; ++l) {
        // ln1 -> xn (f32 for conv) + xnbf (GEMM operand)
        layernorm_k<<<BS_, 256, 0, stream>>>(h, ln1_w + (size_t)l * H_,
                                             xn, xnbf, 0, 0);
        // causal conv + silu -> xcbf
        conv_silu_k<<<((size_t)BS_ * H_) / 256, 256, 0, stream>>>(
            xn, conv_w + (size_t)l * H_ * KC_, conv_b + (size_t)l * H_, xcbf);
        // recurrent weights -> bf16 [4H,H]
        r_transpose_bf16_k<<<((size_t)4 * H_ * H_) / 256, 256, 0, stream>>>(
            R + (size_t)l * H_ * 4 * H_, Rt);
        // gate projections (cell bias folded in): Wx[:, g*H:(g+1)*H]
        for (int g = 0; g < 4; ++g) {
            const __bf16* Ain = (g < 2) ? xcbf : xnbf;
            launch_gemm(Ain, H_, gwbf[l][g],
                        cell_b + (size_t)l * 4 * H_ + g * H_, nullptr, 0,
                        Wx + (size_t)g * H_, 4 * H_, BS_, H_, H_, stream);
        }
        // zero recurrent state + y0
        hipMemsetAsync(cst, 0, (size_t)B_ * H_ * sizeof(float), stream);
        hipMemsetAsync(nst, 0, (size_t)B_ * H_ * sizeof(float), stream);
        hipMemsetAsync(mst, 0, (size_t)B_ * H_ * sizeof(float), stream);
        hipMemsetAsync(ybz, 0, (size_t)B_ * H_ * sizeof(__bf16), stream);
        // sequential recurrence
        dim3 sg(H_ / 16, B_ / 32);
        __bf16* ybuf[2] = { ybf0, ybf1 };
        for (int s = 0; s < S_; ++s) {
            const __bf16* yp = (s == 0) ? ybz : ybuf[(s - 1) & 1];
            slstm_step_k<<<sg, 128, 0, stream>>>(Wx, Rt, yp,
                                                 ys + (size_t)s * B_ * H_,
                                                 ybuf[s & 1],
                                                 cst, nst, mst, s);
        }
        // h += LN(ys)  (group norm, nh=1)
        layernorm_k<<<BS_, 256, 0, stream>>>(ys, gn_w + (size_t)l * H_,
                                             h, nullptr, 1, 1);
        // ln2 -> xn/xnbf (fn)
        layernorm_k<<<BS_, 256, 0, stream>>>(h, ln2_w + (size_t)l * H_,
                                             xn, xnbf, 0, 0);
        // FFN up: gv = fn @ ff_up^T  [BS,2U]
        launch_gemm(xnbf, H_, upbf[l], nullptr, nullptr, 0,
                    gv, 2 * U_, BS_, 2 * U_, H_, stream);
        // gelu(g)*v -> tgbf
        gelu_gate_k<<<((size_t)BS_ * U_) / 256, 256, 0, stream>>>(gv, tgbf);
        // h += tg @ ff_down^T
        launch_gemm(tgbf, U_, dnbf[l], nullptr, h, H_,
                    h, H_, BS_, H_, U_, stream);
    }

    // post-LN on last tokens -> hnbf
    layernorm_k<<<B_, 256, 0, stream>>>(h, post_w, hn, hnbf, 2, 0);
    // out = hn @ fc_w^T + fc_b
    launch_gemm(hnbf, H_, fcw_bf, fc_b, nullptr, 0, out, O_, B_, O_, H_, stream);
}
